// Attention_75402445849315
// MI455X (gfx1250) — matrix operand order repaired and checked by reading, not compiled, not run on hardware
//
#include <hip/hip_runtime.h>
#include <hip/hip_bf16.h>

// MI455X / gfx1250, wave32. GEMMs via v_wmma_f32_16x16x32_f16;
// K/V tiles staged to LDS by the Tensor Data Mover (tensor_load_to_lds).
typedef __attribute__((ext_vector_type(16))) _Float16 v16h;
typedef __attribute__((ext_vector_type(8)))  _Float16 v8h;
typedef __attribute__((ext_vector_type(8)))  float    v8f;
typedef __attribute__((ext_vector_type(4)))  unsigned int u32x4;
typedef __attribute__((ext_vector_type(8)))  int i32x8;
typedef __attribute__((ext_vector_type(4)))  int i32x4;

#define BB 4
#define LL 2048
#define DD 1024
#define HH 16
#define HD 64
#define NJ 32            // hd/2 rotary freqs
#define KT 128           // keys per TDM tile in kernel 2

union H16 { v16h v; v8h h[2]; };

// ---- TDM 2-D tile load: build D# (ISA ch.8) and issue ------------------
// Group0: [1:0]=count=1, [63:32]=lds_addr, [120:64]=global_addr, [127:126]=2
// Group1: [17:16]=data_size(1=2B), tensor_dim0/1, tile_dim0/1, dim0 stride
__device__ __forceinline__ void tdm_load_2d(unsigned lds_off, const void* gaddr,
                                            unsigned tensor_d0, unsigned tensor_d1,
                                            unsigned tile_d0, unsigned tile_d1,
                                            unsigned stride0) {
  const unsigned long long ga = (unsigned long long)gaddr;
  u32x4 g0;
  g0[0] = 1u;                                           // count=1, user mode
  g0[1] = lds_off;                                      // LDS byte address
  g0[2] = (unsigned)(ga & 0xFFFFFFFFu);                 // global_addr[31:0]
  g0[3] = (unsigned)((ga >> 32) & 0x01FFFFFFu) | (2u << 30);  // [56:32]|type=2
  i32x8 g1;
  g1[0] = (int)(1u << 16);                              // data_size=1 (2B)
  g1[1] = (int)((tensor_d0 & 0xFFFFu) << 16);           // tensor_dim0[15:0]
  g1[2] = (int)(((tensor_d0 >> 16) & 0xFFFFu) |
                ((tensor_d1 & 0xFFFFu) << 16));         // d0[31:16] | d1[15:0]
  g1[3] = (int)(((tensor_d1 >> 16) & 0xFFFFu) |
                ((tile_d0 & 0xFFFFu) << 16));           // d1[31:16] | tile_d0
  g1[4] = (int)(tile_d1 & 0xFFFFu);                     // tile_d1, tile_d2=0
  g1[5] = (int)stride0;                                 // dim0 stride [31:0]
  g1[6] = 0;                                            // stride hi / dim1 stride
  g1[7] = 0;
  const i32x4 z4 = {0, 0, 0, 0};                        // 2-D: dims 3/4 unused
#if defined(__clang_major__) && (__clang_major__ >= 23)
  const i32x8 z8 = {0, 0, 0, 0, 0, 0, 0, 0};
  __builtin_amdgcn_tensor_load_to_lds(g0, g1, z4, z4, z8, 0);
#else
  __builtin_amdgcn_tensor_load_to_lds(g0, g1, z4, z4, 0);
#endif
}

// ---- fragment builders -------------------------------------------------
// A-matrix 16x32 f16 layout (ISA 7.12.2): lane lo = row; halves 0..7 hold
// K = hi*8 + h, halves 8..15 hold K = 16 + hi*8 + h.
__device__ __forceinline__ v16h cvt16(float4 a, float4 b, float4 c, float4 d) {
  v16h r;
  r[0]=(_Float16)a.x;  r[1]=(_Float16)a.y;  r[2]=(_Float16)a.z;  r[3]=(_Float16)a.w;
  r[4]=(_Float16)b.x;  r[5]=(_Float16)b.y;  r[6]=(_Float16)b.z;  r[7]=(_Float16)b.w;
  r[8]=(_Float16)c.x;  r[9]=(_Float16)c.y;  r[10]=(_Float16)c.z; r[11]=(_Float16)c.w;
  r[12]=(_Float16)d.x; r[13]=(_Float16)d.y; r[14]=(_Float16)d.z; r[15]=(_Float16)d.w;
  return r;
}
__device__ __forceinline__ v16h load_a_f32(const float* row, int k0, int hi) {
  const float4* p0 = (const float4*)(row + k0 + hi * 8);
  const float4* p1 = (const float4*)(row + k0 + 16 + hi * 8);
  return cvt16(p0[0], p0[1], p1[0], p1[1]);
}
// B-matrix 32x16 f16: lane lo = column n; halves 0..7 hold K = k0 + hi*8 + h,
// halves 8..15 hold K = k0 + 16 + hi*8 + h.
__device__ __forceinline__ v16h load_b_f32(const float* row, int k0, int hi) {
  const float4* p0 = (const float4*)(row + k0 + hi * 8);
  const float4* p1 = (const float4*)(row + k0 + 16 + hi * 8);
  return cvt16(p0[0], p0[1], p1[0], p1[1]);
}

__device__ __forceinline__ float rowmax16(float v) {
  v = fmaxf(v, __shfl_xor(v, 1, 32));
  v = fmaxf(v, __shfl_xor(v, 2, 32));
  v = fmaxf(v, __shfl_xor(v, 4, 32));
  v = fmaxf(v, __shfl_xor(v, 8, 32));
  return v;
}
__device__ __forceinline__ float rowsum16(float v) {
  v += __shfl_xor(v, 1, 32);
  v += __shfl_xor(v, 2, 32);
  v += __shfl_xor(v, 4, 32);
  v += __shfl_xor(v, 8, 32);
  return v;
}

// ---- kernel 1: fused QKV projection + RoPE, f32 -> f16 staging ---------
// grid (BL/64, 3D/32), 256 threads = 8 waves, each wave one 16x16 tile.
__global__ __launch_bounds__(256) void qkv_rope_kernel(
    const float* __restrict__ x,
    const float* __restrict__ Wq, const float* __restrict__ Wk,
    const float* __restrict__ Wv,
    const float* __restrict__ fcos, const float* __restrict__ fsin,
    _Float16* __restrict__ Qh, _Float16* __restrict__ Kh,
    _Float16* __restrict__ Vt) {
  __shared__ _Float16 tile[64 * 32];              // 4 KB output staging
  const int lane = threadIdx.x & 31;
  const int wave = threadIdx.x >> 5;
  const int hi = lane >> 4, lo = lane & 15;
  const int mt = wave & 3, nt = wave >> 2;

  const int mBlk  = blockIdx.x * 64;              // block token base
  const int bIdx  = mBlk / LL;                    // uniform: 64 | 2048
  const int posBlk = mBlk % LL;
  const int mBase = mBlk + mt * 16;
  const int nGlob = blockIdx.y * 32 + nt * 16;    // in [0, 3*D)
  const int proj  = (blockIdx.y * 32) / DD;       // 0=Q 1=K 2=V (uniform)
  const int feat  = nGlob % DD;
  const float* W  = (proj == 0) ? Wq : ((proj == 1) ? Wk : Wv);

  const float* xrow = x + (size_t)(mBase + lo) * DD;  // A lane = row
  const float* wrow = W + (size_t)(feat + lo) * DD;   // B lane = out feature

  v8f acc = {};
  for (int k0 = 0; k0 < DD; k0 += 32) {
    __builtin_prefetch(xrow + ((k0 + 128) & (DD - 1)), 0, 3);
    __builtin_prefetch(wrow + ((k0 + 128) & (DD - 1)), 0, 3);
    v16h a = load_a_f32(xrow, k0, hi);
    v16h b = load_b_f32(wrow, k0, hi);
    acc = __builtin_amdgcn_wmma_f32_16x16x32_f16(false, a, false, b,
                                                 (short)0, acc, false, false);
  }

  // D-fragment: lane lo = column, VGPR r = row r + hi*8.
  const int featLane = feat + lo;
  const int head = featLane >> 6;
  const int d    = featLane & 63;
  const int j    = d >> 1;
  const bool even = (d & 1) == 0;
  const int posW = posBlk + mt * 16 + hi * 8;     // wave row base (r = +0..7)

  if (proj < 2) {
    // RoPE in-register (pairs are adjacent lanes), stage via LDS for
    // coalesced b128 stores.
    for (int r = 0; r < 8; ++r) {
      float v = acc[r];
      const float part = __shfl_xor(v, 1, 32);
      const int pos = posW + r;
      const float c = fcos[pos * NJ + j];
      const float s = fsin[pos * NJ + j];
      v = even ? (v * c - part * s) : (part * s + v * c);
      tile[(mt * 16 + r + hi * 8) * 32 + nt * 16 + lo] = (_Float16)v;
    }
    __syncthreads();
    // cooperative row-major store: 1 x b128 per thread
    _Float16* dst = (proj == 0) ? Qh : Kh;
    const int tok = threadIdx.x >> 2;             // 0..63
    const int seg = threadIdx.x & 3;              // 0..3 (8 halves each)
    const int featBlk = (blockIdx.y * 32) % DD;
    const int headB = featBlk >> 6;
    const int dB    = (featBlk & 63) + seg * 8;
    _Float16* p = dst + (((size_t)bIdx * HH + headB) * LL + posBlk + tok) * HD + dB;
    *(v8h*)p = *(const v8h*)(tile + tok * 32 + seg * 8);
  } else {
    // V transposed [B,H,hd,L]: the 8 fragment rows are 8 consecutive
    // tokens at fixed dim -> single 16-byte store per lane.
    v8h v;
    for (int r = 0; r < 8; ++r) v[r] = (_Float16)acc[r];
    _Float16* p = Vt + (((size_t)bIdx * HH + head) * HD + d) * LL + posW;
    *(v8h*)p = v;
  }
}

// ---- kernel 2: flash attention; TDM stages K/V tiles to LDS ------------
// grid (L/128, B*H), 8 waves x 16 query rows; 128-key TDM tiles.
__global__ __launch_bounds__(256) void flash_attn_kernel(
    const _Float16* __restrict__ Qh, const _Float16* __restrict__ Kh,
    const _Float16* __restrict__ Vt, const float* __restrict__ mask,
    _Float16* __restrict__ Oh) {
  __shared__ _Float16 sh[8 * 16 * 64];     // P/O staging (per-wave 1024 halves)
  __shared__ _Float16 shK[KT * HD];        // K tile: 128 keys x 64 dims, 16 KB
  __shared__ _Float16 shV[HD * KT];        // V tile: 64 dims x 128 keys, 16 KB
  const int lane = threadIdx.x & 31;
  const int wave = threadIdx.x >> 5;
  const int hi = lane >> 4, lo = lane & 15;
  const int bh = blockIdx.y;
  const int bIdx = bh / HH, head = bh % HH;
  const int qBase = blockIdx.x * 128 + wave * 16;

  // Q A-fragments (2 x 16x32 covering hd=64), pre-scaled by 1/sqrt(hd)
  const _Float16* qrow = Qh + ((size_t)bh * LL + qBase + lo) * HD;
  v16h qa[2];
  for (int t = 0; t < 2; ++t) {
    H16 u;
    u.h[0] = *(const v8h*)(qrow + t * 32 + hi * 16);
    u.h[1] = *(const v8h*)(qrow + t * 32 + hi * 16 + 8);
    for (int i = 0; i < 16; ++i) u.v[i] = u.v[i] * (_Float16)0.125f;
    qa[t] = u.v;
  }

  v8f o0 = {}, o1 = {}, o2 = {}, o3 = {};
  float mrow[8], lrow[8];
  for (int r = 0; r < 8; ++r) { mrow[r] = -1e30f; lrow[r] = 0.f; }

  _Float16* lp = sh + wave * 1024;
  const _Float16* kbase = Kh + (size_t)bh * LL * HD;
  const _Float16* vbase = Vt + (size_t)bh * HD * LL;
  const float*    mptr  = mask + (size_t)bIdx * LL;
  const unsigned ldsK = (unsigned)(uintptr_t)(void*)shK;
  const unsigned ldsV = (unsigned)(uintptr_t)(void*)shV;

  for (int kt = 0; kt < LL; kt += KT) {
    // ---- TDM: DMA this block's K and V tiles into LDS (wave 0 issues) ----
    if (wave == 0) {
      // K tile: 128 rows (keys) x 64 halves, row stride HD
      tdm_load_2d(ldsK, kbase + (size_t)kt * HD, HD, LL, HD, KT, HD);
      // V tile: 64 rows (dims) x 128 halves (keys), row stride LL
      tdm_load_2d(ldsV, vbase + kt, LL, HD, KT, HD, LL);
      __builtin_amdgcn_s_wait_tensorcnt(0);
    }
    __syncthreads();

    for (int n0 = 0; n0 < KT; n0 += 32) {
      // ---- S = Q K^T from LDS (4 WMMAs) ----
      v8f s[2];
      for (int st = 0; st < 2; ++st) {
        const _Float16* krow = shK + (n0 + st * 16 + lo) * HD;
        v8f acc = {};
        v16h b0 = *(const v16h*)(krow + hi * 16);
        acc = __builtin_amdgcn_wmma_f32_16x16x32_f16(false, qa[0], false, b0,
                                                     (short)0, acc, false, false);
        v16h b1 = *(const v16h*)(krow + 32 + hi * 16);
        acc = __builtin_amdgcn_wmma_f32_16x16x32_f16(false, qa[1], false, b1,
                                                     (short)0, acc, false, false);
        const float mv = mptr[kt + n0 + st * 16 + lo];  // additive key mask
        for (int i = 0; i < 8; ++i) acc[i] += mv;
        s[st] = acc;
      }

      // ---- online softmax; stash P (f16) into LDS in A-fragment order ----
      for (int r = 0; r < 8; ++r) {
        float mx = rowmax16(fmaxf(s[0][r], s[1][r]));
        const float mn   = fmaxf(mrow[r], mx);
        const float corr = __expf(mrow[r] - mn);
        const float p0 = __expf(s[0][r] - mn);
        const float p1 = __expf(s[1][r] - mn);
        const float rs = rowsum16(p0 + p1);
        lrow[r] = lrow[r] * corr + rs;
        mrow[r] = mn;
        o0[r] *= corr; o1[r] *= corr; o2[r] *= corr; o3[r] *= corr;
        const int row = r + hi * 8;
        lp[row * 32 + lo]      = (_Float16)p0;
        lp[row * 32 + 16 + lo] = (_Float16)p1;
      }

      // C-fragment -> A-fragment re-layout through LDS (ds_load_b128)
      const _Float16* prow = lp + lo * 32;
      H16 u;
      u.h[0] = *(const v8h*)(prow + hi * 16);
      u.h[1] = *(const v8h*)(prow + hi * 16 + 8);
      const v16h pa = u.v;

      // ---- O += P V from LDS (4 WMMAs over hd=64) ----
      v16h vb;
      vb = *(const v16h*)(shV + (0 * 16 + lo) * KT + n0 + hi * 16);
      o0 = __builtin_amdgcn_wmma_f32_16x16x32_f16(false, pa, false, vb, (short)0, o0, false, false);
      vb = *(const v16h*)(shV + (1 * 16 + lo) * KT + n0 + hi * 16);
      o1 = __builtin_amdgcn_wmma_f32_16x16x32_f16(false, pa, false, vb, (short)0, o1, false, false);
      vb = *(const v16h*)(shV + (2 * 16 + lo) * KT + n0 + hi * 16);
      o2 = __builtin_amdgcn_wmma_f32_16x16x32_f16(false, pa, false, vb, (short)0, o2, false, false);
      vb = *(const v16h*)(shV + (3 * 16 + lo) * KT + n0 + hi * 16);
      o3 = __builtin_amdgcn_wmma_f32_16x16x32_f16(false, pa, false, vb, (short)0, o3, false, false);
    }
    __syncthreads();   // tile consumed; safe to overwrite next iteration
  }

  // normalize into wave-private LDS staging tile (16x64), then store the
  // whole tile as coalesced b128 (4 per lane).
  for (int r = 0; r < 8; ++r) {
    const float inv = 1.0f / lrow[r];
    const int row = r + hi * 8;
    lp[row * 64 + lo]      = (_Float16)(o0[r] * inv);
    lp[row * 64 + 16 + lo] = (_Float16)(o1[r] * inv);
    lp[row * 64 + 32 + lo] = (_Float16)(o2[r] * inv);
    lp[row * 64 + 48 + lo] = (_Float16)(o3[r] * inv);
  }
  _Float16* obase = Oh + ((size_t)bIdx * LL + qBase) * DD + head * HD;
  for (int sgi = 0; sgi < 4; ++sgi) {
    const int idx = sgi * 32 + lane;              // 0..127 segments
    const int row = idx >> 3;                     // 0..15
    const int seg = idx & 7;                      // 0..7 (8 halves each)
    *(v8h*)(obase + (size_t)row * DD + seg * 8) =
        *(const v8h*)(lp + row * 64 + seg * 8);
  }
}

// ---- kernel 3: output projection O @ Wo^T -> f32 ----------------------
__global__ __launch_bounds__(256) void out_proj_kernel(
    const _Float16* __restrict__ Oh, const float* __restrict__ Wo,
    float* __restrict__ out) {
  __shared__ float tile[64 * 32];                 // 8 KB output staging
  const int lane = threadIdx.x & 31;
  const int wave = threadIdx.x >> 5;
  const int hi = lane >> 4, lo = lane & 15;
  const int mt = wave & 3, nt = wave >> 2;
  const int mBase = blockIdx.x * 64 + mt * 16;
  const int nBase = blockIdx.y * 32 + nt * 16;

  const _Float16* arow = Oh + (size_t)(mBase + lo) * DD;
  const float*    wrow = Wo + (size_t)(nBase + lo) * DD;

  v8f acc = {};
  for (int k0 = 0; k0 < DD; k0 += 32) {
    __builtin_prefetch(wrow + ((k0 + 128) & (DD - 1)), 0, 3);
    H16 u;
    u.h[0] = *(const v8h*)(arow + k0 + hi * 8);
    u.h[1] = *(const v8h*)(arow + k0 + 16 + hi * 8);
    v16h b = load_b_f32(wrow, k0, hi);
    acc = __builtin_amdgcn_wmma_f32_16x16x32_f16(false, u.v, false, b,
                                                 (short)0, acc, false, false);
  }
  for (int r = 0; r < 8; ++r)
    tile[(mt * 16 + r + hi * 8) * 32 + nt * 16 + lo] = acc[r];
  __syncthreads();

  // coalesced f32 stores: 2 x b128 per thread
  const int tok = threadIdx.x >> 2;               // 0..63
  const int seg = threadIdx.x & 3;                // 0..3 (8 floats each)
  float* p = out + (size_t)(blockIdx.x * 64 + tok) * DD + blockIdx.y * 32 + seg * 8;
  const float* q = tile + tok * 32 + seg * 8;
  *(float4*)p       = *(const float4*)q;
  *(float4*)(p + 4) = *(const float4*)(q + 4);
}

// ---- launcher ----------------------------------------------------------
extern "C" void kernel_launch(void* const* d_in, const int* in_sizes, int n_in,
                              void* d_out, int out_size, void* d_ws, size_t ws_size,
                              hipStream_t stream) {
  const float* x    = (const float*)d_in[0];
  const float* fcos = (const float*)d_in[1];
  const float* fsin = (const float*)d_in[2];
  const float* mask = (const float*)d_in[3];
  const float* Wq   = (const float*)d_in[4];
  const float* Wk   = (const float*)d_in[5];
  const float* Wv   = (const float*)d_in[6];
  const float* Wo   = (const float*)d_in[7];
  float* out = (float*)d_out;

  // f16 staging: Q, K, V^T, O -- 4 x 16 MB in workspace (L2-resident)
  const size_t elems = (size_t)BB * LL * DD;
  _Float16* Qh = (_Float16*)d_ws;
  _Float16* Kh = Qh + elems;
  _Float16* Vt = Kh + elems;
  _Float16* Oh = Vt + elems;

  qkv_rope_kernel<<<dim3((BB * LL) / 64, (3 * DD) / 32), 256, 0, stream>>>(
      x, Wq, Wk, Wv, fcos, fsin, Qh, Kh, Vt);
  flash_attn_kernel<<<dim3(LL / 128, BB * HH), 256, 0, stream>>>(
      Qh, Kh, Vt, mask, Oh);
  out_proj_kernel<<<dim3((BB * LL) / 64, DD / 32), 256, 0, stream>>>(
      Oh, Wo, out);
}
